// DecoderLayer_23562190586281
// MI455X (gfx1250) — matrix operand order repaired and checked by reading, not compiled, not run on hardware
//
#include <hip/hip_runtime.h>
#include <hip/hip_bf16.h>

// ---------------------------------------------------------------------------
// Decoder layer for MI455X (gfx1250): bf16 WMMA GEMMs + flash attention,
// with CDNA5 async global->LDS copies (ASYNCcnt) double-buffering the GEMM.
// ---------------------------------------------------------------------------

typedef __bf16 v16bf __attribute__((ext_vector_type(16)));
typedef __bf16 v8bf  __attribute__((ext_vector_type(8)));
typedef float  v8f   __attribute__((ext_vector_type(8)));

#define DMODEL 1024
#define NHEAD  16
#define DHEAD  64
#define DFFN   4096
#define LQ_    2048
#define LKV_   4096
#define B_     2

__device__ __forceinline__ unsigned short f2bf(float f) {
  unsigned int u = __float_as_uint(f);
  unsigned int r = (u + 0x7FFFu + ((u >> 16) & 1u)) >> 16;  // RNE
  return (unsigned short)r;
}

__device__ __forceinline__ v8f wmma_bf16(v16bf a, v16bf b, v8f c) {
  return __builtin_amdgcn_wmma_f32_16x16x32_bf16(false, a, false, b, (short)0, c,
                                                 false, false);
}

// CDNA5 async global->LDS copy, GV mode (64-bit per-lane vaddr, SADDR=off).
// Per ISA 08, the copy reads global vaddr+offset and writes LDS at vdst+offset,
// i.e. the instruction offset applies to BOTH addresses. Tracked by ASYNCcnt.
__device__ __forceinline__ void async_ld_b128(unsigned lds_addr, const void* gptr,
                                              int ioff) {
  asm volatile("global_load_async_to_lds_b128 %0, %1, off offset:%2"
               :
               : "v"(lds_addr), "v"(gptr), "n"(ioff)
               : "memory");
}

template <int N>
__device__ __forceinline__ void wait_async() {
  asm volatile("s_wait_asynccnt %0" : : "n"(N) : "memory");
}

// Low 32 bits of a generic pointer into LDS == LDS byte offset on AMDGCN.
__device__ __forceinline__ unsigned lds_off(const void* p) {
  return (unsigned)(unsigned long long)p;
}

// A-fragment (16x32 bf16), ISA layout: lane<16 -> M=lane, K {0..7,16..23};
// lane>=16 -> M=lane-16, K {8..15,24..31}. Two 16B contiguous LDS reads.
__device__ __forceinline__ v16bf frag_a(const unsigned short* tile, int stride) {
  int lane = threadIdx.x & 31;
  int m = lane & 15, kh = lane >> 4;
  const unsigned short* p = tile + m * stride + kh * 8;
  v8bf lo = *(const v8bf*)p;
  v8bf hi = *(const v8bf*)(p + 16);
  return __builtin_shufflevector(lo, hi, 0, 1, 2, 3, 4, 5, 6, 7, 8, 9, 10, 11,
                                 12, 13, 14, 15);
}

// B-fragment (32x16 bf16): lane<16 -> N=lane, K {0..7,16..23}; lane>=16 -> K {8..15,24..31}.
__device__ __forceinline__ v16bf frag_b(const unsigned short* tile, int stride) {
  int lane = threadIdx.x & 31;
  int n = lane & 15, kh = lane >> 4;
  const unsigned short* p = tile + n * stride + kh * 8;
  v8bf lo = *(const v8bf*)p;
  v8bf hi = *(const v8bf*)(p + 16);
  return __builtin_shufflevector(lo, hi, 0, 1, 2, 3, 4, 5, 6, 7, 8, 9, 10, 11,
                                 12, 13, 14, 15);
}

// --------------------------- fp32 -> bf16 ----------------------------------
__global__ void k_f32_to_bf16(const float* __restrict__ s,
                              unsigned short* __restrict__ d, int n) {
  int i = blockIdx.x * blockDim.x + threadIdx.x;
  int st = gridDim.x * blockDim.x;
  for (; i < n; i += st) d[i] = f2bf(s[i]);
}

// ------------------------------ LayerNorm ----------------------------------
__global__ __launch_bounds__(256) void k_layernorm(
    const float* __restrict__ x, const float* __restrict__ g,
    const float* __restrict__ bt, float* __restrict__ yf,
    unsigned short* __restrict__ yb) {
  int row = blockIdx.x;
  int t = threadIdx.x;
  const float* xr = x + (size_t)row * DMODEL;
  float v[4];
  float s = 0.f, sq = 0.f;
#pragma unroll
  for (int i = 0; i < 4; ++i) {
    float a = xr[t + i * 256];
    v[i] = a;
    s += a;
    sq += a * a;
  }
#pragma unroll
  for (int off = 16; off; off >>= 1) {
    s += __shfl_xor(s, off);
    sq += __shfl_xor(sq, off);
  }
  __shared__ float ss[8], ssq[8], mu_s, rs_s;
  int wid = t >> 5;
  if ((t & 31) == 0) { ss[wid] = s; ssq[wid] = sq; }
  __syncthreads();
  if (t == 0) {
    float S = 0.f, Q = 0.f;
    for (int i = 0; i < 8; ++i) { S += ss[i]; Q += ssq[i]; }
    float mu = S * (1.f / DMODEL);
    float var = Q * (1.f / DMODEL) - mu * mu;
    mu_s = mu;
    rs_s = rsqrtf(var + 1e-5f);
  }
  __syncthreads();
  float mu = mu_s, rs = rs_s;
#pragma unroll
  for (int i = 0; i < 4; ++i) {
    int c = t + i * 256;
    float y = (v[i] - mu) * rs * g[c] + bt[c];
    if (yf) yf[(size_t)row * DMODEL + c] = y;
    yb[(size_t)row * DMODEL + c] = f2bf(y);
  }
}

// ------------------------------- GEMM --------------------------------------
// C[M,N] = A[M,K] @ W[N,K]^T.  128x128 block tile, K-step 32, double-buffered
// LDS filled by async global->LDS b128 copies overlapped with WMMA compute.
template <bool BIAS, bool RELU, bool RES, bool OUTF, bool OUTB>
__global__ __launch_bounds__(256) void k_gemm_bf16(
    const unsigned short* __restrict__ A, const unsigned short* __restrict__ W,
    const float* __restrict__ bias, const float* __restrict__ resid,
    float* __restrict__ outf, unsigned short* __restrict__ outb, int M, int N,
    int K) {
  __shared__ unsigned short As[2][128 * 40];  // padded stride 40
  __shared__ unsigned short Ws[2][128 * 40];
  int t = threadIdx.x;
  int lane = t & 31, wid = t >> 5;
  int wm = (wid & 3) * 32, wn = (wid >> 2) * 64;
  int bm = blockIdx.y * 128, bn = blockIdx.x * 128;

  v8f acc[2][4];
#pragma unroll
  for (int mi = 0; mi < 2; ++mi)
#pragma unroll
    for (int ni = 0; ni < 4; ++ni)
#pragma unroll
      for (int e = 0; e < 8; ++e) acc[mi][ni][e] = 0.f;

  int lr = t >> 1, lc = (t & 1) * 16;  // 2 threads/row, 32B (= 2x b128) each
  const unsigned short* gA = A + (size_t)(bm + lr) * K + lc;
  const unsigned short* gW = W + (size_t)(bn + lr) * K + lc;
  unsigned lAs0 = lds_off(&As[0][lr * 40 + lc]);
  unsigned lAs1 = lds_off(&As[1][lr * 40 + lc]);
  unsigned lWs0 = lds_off(&Ws[0][lr * 40 + lc]);
  unsigned lWs1 = lds_off(&Ws[1][lr * 40 + lc]);

  // prologue: tile 0 -> buffer 0
  async_ld_b128(lAs0, gA, 0);
  async_ld_b128(lAs0, gA, 16);
  async_ld_b128(lWs0, gW, 0);
  async_ld_b128(lWs0, gW, 16);

  int buf = 0;
  for (int k0 = 0; k0 < K; k0 += 32) {
    bool more = (k0 + 32) < K;
    if (more) {  // prefetch tile k0+32 into the other buffer
      unsigned la = buf ? lAs0 : lAs1;
      unsigned lw = buf ? lWs0 : lWs1;
      const unsigned short* pa = gA + k0 + 32;
      const unsigned short* pw = gW + k0 + 32;
      async_ld_b128(la, pa, 0);
      async_ld_b128(la, pa, 16);
      async_ld_b128(lw, pw, 0);
      async_ld_b128(lw, pw, 16);
      wait_async<4>();  // current tile done; only prefetch in flight
    } else {
      wait_async<0>();
    }
    __syncthreads();  // all waves' async fills visible

    const unsigned short* at = As[buf];
    const unsigned short* wt = Ws[buf];
    v16bf af[2], bf[4];
#pragma unroll
    for (int mi = 0; mi < 2; ++mi) af[mi] = frag_a(&at[(wm + mi * 16) * 40], 40);
#pragma unroll
    for (int ni = 0; ni < 4; ++ni) bf[ni] = frag_b(&wt[(wn + ni * 16) * 40], 40);
#pragma unroll
    for (int mi = 0; mi < 2; ++mi)
#pragma unroll
      for (int ni = 0; ni < 4; ++ni)
        acc[mi][ni] = wmma_bf16(af[mi], bf[ni], acc[mi][ni]);

    __syncthreads();  // buffer free before it is re-issued two tiles later
    buf ^= 1;
  }

  int kh = lane >> 4, nl = lane & 15;
#pragma unroll
  for (int mi = 0; mi < 2; ++mi) {
#pragma unroll
    for (int ni = 0; ni < 4; ++ni) {
      int col = bn + wn + ni * 16 + nl;
      float bv = BIAS ? bias[col] : 0.f;
#pragma unroll
      for (int r = 0; r < 8; ++r) {
        int row = bm + wm + mi * 16 + r + 8 * kh;
        float val = acc[mi][ni][r] + bv;
        if (RELU) val = fmaxf(val, 0.f);
        if (RES) val += resid[(size_t)row * N + col];
        if (OUTF) outf[(size_t)row * N + col] = val;
        if (OUTB) outb[(size_t)row * N + col] = f2bf(val);
      }
    }
  }
}

// --------------------------- Flash attention -------------------------------
// Per block: one (batch, head), 128 q rows; 8 waves x 16 q rows each.
// K chunk filled by async global->LDS, overlapped with the V transpose.
__global__ __launch_bounds__(256) void k_flash_attn(
    const unsigned short* __restrict__ Qb, const unsigned short* __restrict__ Kb,
    const unsigned short* __restrict__ Vb, unsigned short* __restrict__ Ob,
    int Lq, int Lkv, float scale) {
  __shared__ unsigned short Qs[128 * 72];      // [q][d]
  __shared__ unsigned short Ks[64 * 72];       // [kv][d]
  __shared__ unsigned short Vs[64 * 72];       // transposed: [d][kv]
  __shared__ unsigned short Ps[8 * 16 * 72];   // per-wave P scratch [q][kv]
  int t = threadIdx.x;
  int lane = t & 31, wid = t >> 5;
  int bh = blockIdx.x;
  int b = bh / NHEAD, h = bh % NHEAD;
  int q0 = blockIdx.y * 128;
  int co = h * DHEAD;

  {  // load Q tile (128 x 64 bf16)
    int r = t >> 1, c = (t & 1) * 32;
    const uint4* src =
        (const uint4*)(Qb + (size_t)(b * Lq + q0 + r) * DMODEL + co + c);
    uint4 a0 = src[0], a1 = src[1], a2 = src[2], a3 = src[3];
    *(uint4*)&Qs[r * 72 + c] = a0;
    *(uint4*)&Qs[r * 72 + c + 8] = a1;
    *(uint4*)&Qs[r * 72 + c + 16] = a2;
    *(uint4*)&Qs[r * 72 + c + 24] = a3;
  }

  float mrun[8], lrun[8];
  v8f oacc[4];
#pragma unroll
  for (int r = 0; r < 8; ++r) { mrun[r] = -3.0e38f; lrun[r] = 0.f; }
#pragma unroll
  for (int d = 0; d < 4; ++d)
#pragma unroll
    for (int e = 0; e < 8; ++e) oacc[d][e] = 0.f;

  int wq = wid * 16;
  int kh = lane >> 4, nl = lane & 15;
  unsigned short* Pw = &Ps[wid * 16 * 72];

  int r4 = t >> 2, c4 = (t & 3) * 16;
  unsigned lK = lds_off(&Ks[r4 * 72 + c4]);

  for (int c0 = 0; c0 < Lkv; c0 += 64) {
    __syncthreads();  // previous chunk fully consumed
    {
      // K chunk: async global->LDS (32B per thread via two b128 ops)
      const unsigned short* gk =
          Kb + (size_t)(b * Lkv + c0 + r4) * DMODEL + co + c4;
      async_ld_b128(lK, gk, 0);
      async_ld_b128(lK, gk, 16);
      // V chunk: load + transpose into [d][kv] (contiguous B-fragments)
      const uint4* sv =
          (const uint4*)(Vb + (size_t)(b * Lkv + c0 + r4) * DMODEL + co + c4);
      uint4 v0v = sv[0], v1v = sv[1];
      const unsigned short* e0 = (const unsigned short*)&v0v;
      const unsigned short* e1 = (const unsigned short*)&v1v;
#pragma unroll
      for (int i = 0; i < 8; ++i) Vs[(c4 + i) * 72 + r4] = e0[i];
#pragma unroll
      for (int i = 0; i < 8; ++i) Vs[(c4 + 8 + i) * 72 + r4] = e1[i];
      wait_async<0>();
    }
    __syncthreads();

    // S = Q @ K^T  (16 q rows x 64 kv, per wave)
    v8f s[4];
#pragma unroll
    for (int j = 0; j < 4; ++j)
#pragma unroll
      for (int e = 0; e < 8; ++e) s[j][e] = 0.f;
#pragma unroll
    for (int kt = 0; kt < 64; kt += 32) {
      v16bf a = frag_a(&Qs[wq * 72 + kt], 72);
#pragma unroll
      for (int j = 0; j < 4; ++j) {
        v16bf bfr = frag_b(&Ks[j * 16 * 72 + kt], 72);
        s[j] = wmma_bf16(a, bfr, s[j]);
      }
    }

    // Online softmax: each C-row lives in one 16-lane half -> xor 1,2,4,8.
#pragma unroll
    for (int r = 0; r < 8; ++r) {
      float mx = s[0][r] * scale;
#pragma unroll
      for (int j = 1; j < 4; ++j) mx = fmaxf(mx, s[j][r] * scale);
#pragma unroll
      for (int off = 8; off; off >>= 1) mx = fmaxf(mx, __shfl_xor(mx, off));
      float mr = fmaxf(mrun[r], mx);
      float alpha = __expf(mrun[r] - mr);
      float rs = 0.f;
#pragma unroll
      for (int j = 0; j < 4; ++j) {
        float p = __expf(s[j][r] * scale - mr);
        s[j][r] = p;
        rs += p;
      }
#pragma unroll
      for (int off = 8; off; off >>= 1) rs += __shfl_xor(rs, off);
      lrun[r] = lrun[r] * alpha + rs;
      mrun[r] = mr;
#pragma unroll
      for (int d = 0; d < 4; ++d) oacc[d][r] *= alpha;
    }

    // P -> LDS bf16 (C layout -> row-major), wave-private scratch
#pragma unroll
    for (int j = 0; j < 4; ++j)
#pragma unroll
      for (int r = 0; r < 8; ++r)
        Pw[(r + 8 * kh) * 72 + j * 16 + nl] = f2bf(s[j][r]);

    // O += P @ V
#pragma unroll
    for (int kt = 0; kt < 64; kt += 32) {
      v16bf a = frag_a(&Pw[kt], 72);
#pragma unroll
      for (int d = 0; d < 4; ++d) {
        v16bf bfr = frag_b(&Vs[d * 16 * 72 + kt], 72);
        oacc[d] = wmma_bf16(a, bfr, oacc[d]);
      }
    }
  }

  // O / l -> bf16 output (heads concatenated along feature dim)
#pragma unroll
  for (int d = 0; d < 4; ++d) {
#pragma unroll
    for (int r = 0; r < 8; ++r) {
      int row = b * Lq + q0 + wq + r + 8 * kh;
      float val = oacc[d][r] / lrun[r];
      Ob[(size_t)row * DMODEL + co + d * 16 + nl] = f2bf(val);
    }
  }
}

// ------------------------------- driver ------------------------------------
extern "C" void kernel_launch(void* const* d_in, const int* in_sizes, int n_in,
                              void* d_out, int out_size, void* d_ws,
                              size_t ws_size, hipStream_t stream) {
  (void)in_sizes; (void)n_in; (void)out_size; (void)ws_size;
  const float* q = (const float*)d_in[0];
  const float* k = (const float*)d_in[1];
  const float* v = (const float*)d_in[2];
  const float* wsrc[10] = {
      (const float*)d_in[3], (const float*)d_in[4],  (const float*)d_in[5],
      (const float*)d_in[6], (const float*)d_in[7],  (const float*)d_in[8],
      (const float*)d_in[9], (const float*)d_in[10], (const float*)d_in[21],
      (const float*)d_in[23]};
  const float* g_sa = (const float*)d_in[11];
  const float* b_sa = (const float*)d_in[12];
  const float* g_qca = (const float*)d_in[13];
  const float* b_qca = (const float*)d_in[14];
  const float* g_kca = (const float*)d_in[15];
  const float* b_kca = (const float*)d_in[16];
  const float* g_vca = (const float*)d_in[17];
  const float* b_vca = (const float*)d_in[18];
  const float* g_out = (const float*)d_in[19];
  const float* b_out = (const float*)d_in[20];
  const float* fb1 = (const float*)d_in[22];
  const float* fb2 = (const float*)d_in[24];
  float* out = (float*)d_out;

  const int MQ = B_ * LQ_;    // 4096 query rows
  const int MKV = B_ * LKV_;  // 8192 kv rows

  char* ws = (char*)d_ws;
  size_t off = 0;
  auto alloc = [&](size_t bytes) {
    size_t o = off;
    off += (bytes + 255) & ~(size_t)255;
    return o;
  };
  unsigned short* wb[10];
  const size_t wcnt[10] = {1048576, 1048576, 1048576, 1048576, 1048576,
                           1048576, 1048576, 1048576,
                           (size_t)DFFN * DMODEL, (size_t)DFFN * DMODEL};
  for (int i = 0; i < 10; ++i)
    wb[i] = (unsigned short*)(ws + alloc(wcnt[i] * 2));
  float* q1f = (float*)(ws + alloc((size_t)MQ * DMODEL * 4));
  unsigned short* q1b = (unsigned short*)(ws + alloc((size_t)MQ * DMODEL * 2));
  unsigned short* pq = (unsigned short*)(ws + alloc((size_t)MQ * DMODEL * 2));
  unsigned short* pk = (unsigned short*)(ws + alloc((size_t)MKV * DMODEL * 2));
  unsigned short* pv = (unsigned short*)(ws + alloc((size_t)MKV * DMODEL * 2));
  unsigned short* ao = (unsigned short*)(ws + alloc((size_t)MQ * DMODEL * 2));
  float* q2f = (float*)(ws + alloc((size_t)MQ * DMODEL * 4));
  float* q3f = (float*)(ws + alloc((size_t)MQ * DMODEL * 4));
  unsigned short* q3b = (unsigned short*)(ws + alloc((size_t)MQ * DMODEL * 2));
  unsigned short* k3b = (unsigned short*)(ws + alloc((size_t)MKV * DMODEL * 2));
  unsigned short* v3b = (unsigned short*)(ws + alloc((size_t)MKV * DMODEL * 2));
  float* q4f = (float*)(ws + alloc((size_t)MQ * DMODEL * 4));
  unsigned short* hb = (unsigned short*)(ws + alloc((size_t)MQ * DMODEL * 2));
  unsigned short* h1b = (unsigned short*)(ws + alloc((size_t)MQ * DFFN * 2));

  for (int i = 0; i < 10; ++i)
    k_f32_to_bf16<<<dim3(2048), 256, 0, stream>>>(wsrc[i], wb[i], (int)wcnt[i]);

  dim3 gDD(DMODEL / 128, MQ / 128);
  dim3 gKV(DMODEL / 128, MKV / 128);

  // ---- self-attention block: q1 = LN(q); q2 = q1 + MHA(q1,q1,q1) ----
  k_layernorm<<<MQ, 256, 0, stream>>>(q, g_sa, b_sa, q1f, q1b);
  k_gemm_bf16<false, false, false, false, true><<<gDD, 256, 0, stream>>>(
      q1b, wb[0], nullptr, nullptr, nullptr, pq, MQ, DMODEL, DMODEL);
  k_gemm_bf16<false, false, false, false, true><<<gDD, 256, 0, stream>>>(
      q1b, wb[1], nullptr, nullptr, nullptr, pk, MQ, DMODEL, DMODEL);
  k_gemm_bf16<false, false, false, false, true><<<gDD, 256, 0, stream>>>(
      q1b, wb[2], nullptr, nullptr, nullptr, pv, MQ, DMODEL, DMODEL);
  k_flash_attn<<<dim3(B_ * NHEAD, LQ_ / 128), 256, 0, stream>>>(
      pq, pk, pv, ao, LQ_, LQ_, 0.125f);
  k_gemm_bf16<false, false, true, true, false><<<gDD, 256, 0, stream>>>(
      ao, wb[3], nullptr, q1f, q2f, nullptr, MQ, DMODEL, DMODEL);

  // ---- cross-attention: q3=LN(q2), k3=LN(k), v3=LN(v); q4 = q3 + MHA ----
  k_layernorm<<<MQ, 256, 0, stream>>>(q2f, g_qca, b_qca, q3f, q3b);
  k_layernorm<<<MKV, 256, 0, stream>>>(k, g_kca, b_kca, nullptr, k3b);
  k_layernorm<<<MKV, 256, 0, stream>>>(v, g_vca, b_vca, nullptr, v3b);
  k_gemm_bf16<false, false, false, false, true><<<gDD, 256, 0, stream>>>(
      q3b, wb[4], nullptr, nullptr, nullptr, pq, MQ, DMODEL, DMODEL);
  k_gemm_bf16<false, false, false, false, true><<<gKV, 256, 0, stream>>>(
      k3b, wb[5], nullptr, nullptr, nullptr, pk, MKV, DMODEL, DMODEL);
  k_gemm_bf16<false, false, false, false, true><<<gKV, 256, 0, stream>>>(
      v3b, wb[6], nullptr, nullptr, nullptr, pv, MKV, DMODEL, DMODEL);
  k_flash_attn<<<dim3(B_ * NHEAD, LQ_ / 128), 256, 0, stream>>>(
      pq, pk, pv, ao, LQ_, LKV_, 0.125f);
  k_gemm_bf16<false, false, true, true, false><<<gDD, 256, 0, stream>>>(
      ao, wb[7], nullptr, q3f, q4f, nullptr, MQ, DMODEL, DMODEL);

  // ---- FFN: out = q4 + W2 @ relu(W1 @ LN(q4) + b1) + b2 ----
  k_layernorm<<<MQ, 256, 0, stream>>>(q4f, g_out, b_out, nullptr, hb);
  k_gemm_bf16<true, true, false, false, true>
      <<<dim3(DFFN / 128, MQ / 128), 256, 0, stream>>>(
          hb, wb[8], fb1, nullptr, nullptr, h1b, MQ, DFFN, DMODEL);
  k_gemm_bf16<true, false, true, true, false><<<gDD, 256, 0, stream>>>(
      h1b, wb[9], fb2, q4f, out, nullptr, MQ, DMODEL, DFFN);
}
